// OTMatchingModuleV2_21148418966329
// MI455X (gfx1250) — matrix operand order repaired and checked by reading, not compiled, not run on hardware
//
#include <hip/hip_runtime.h>
#include <hip/hip_bf16.h>
#include <math.h>

// ---------------------------------------------------------------------------
// FGW + Sinkhorn pipeline for MI455X (gfx1250, wave32, WMMA).
// GEMMs: bf16 inputs, f32 accumulate via V_WMMA_F32_16X16X32_BF16, with
// double-buffered async global->LDS staging (ASYNCcnt) when available.
// Sinkhorn/logsumexp: f32, memory-bound, single-pass online LSE.
// ---------------------------------------------------------------------------

typedef __attribute__((ext_vector_type(16))) __bf16 v16bf;
typedef __attribute__((ext_vector_type(8)))  float  v8f;
typedef int v4i_pack __attribute__((ext_vector_type(4)));
typedef __attribute__((address_space(1))) v4i_pack* as1_v4i;
typedef __attribute__((address_space(3))) v4i_pack* as3_v4i;

#define N4K   4096
#define DFEAT 256
#define NM_TOT (4096*4096)
#define EPSREG 0.05f
#define LAMW   0.5f
#define LOG_AB (-8.317766166719343f)   // log(1/4096)
#define KEY_NEG_INF 0x007FFFFFu        // fkey(-inf)

#if defined(__has_builtin)
# if __has_builtin(__builtin_amdgcn_global_load_async_to_lds_b128)
#  define HAVE_ASYNC_LDS 1
# endif
#endif
#ifndef HAVE_ASYNC_LDS
# define HAVE_ASYNC_LDS 0
#endif

// async global->LDS 16B copy (per-lane addresses); sync fallback
static __device__ __forceinline__ void copy16_g2l(const unsigned short* gp, unsigned short* lp){
#if HAVE_ASYNC_LDS
  __builtin_amdgcn_global_load_async_to_lds_b128(
      (as1_v4i)(unsigned short*)gp, (as3_v4i)lp, 0, 0);
#else
  *(uint4*)lp = *(const uint4*)gp;
#endif
}
static __device__ __forceinline__ void asyncs_wait(){
#if HAVE_ASYNC_LDS
# if __has_builtin(__builtin_amdgcn_s_wait_asynccnt)
  __builtin_amdgcn_s_wait_asynccnt(0);
# else
  asm volatile("s_wait_asynccnt 0x0" ::: "memory");
# endif
#endif
}

// monotone float -> uint key so atomicMax(uint) implements float max (any sign)
static __device__ __forceinline__ unsigned fkey(float f){
  unsigned u = __float_as_uint(f);
  return (u & 0x80000000u) ? ~u : (u | 0x80000000u);
}
static __device__ __forceinline__ float funkey(unsigned k){
  return (k & 0x80000000u) ? __uint_as_float(k & 0x7FFFFFFFu)
                           : __uint_as_float(~k);
}
static __device__ __forceinline__ unsigned short f2bf(float f){
  unsigned u = __float_as_uint(f);
  return (unsigned short)((u + 0x7FFFu + ((u >> 16) & 1u)) >> 16); // RNE
}
static __device__ __forceinline__ float bf2f(unsigned short s){
  return __uint_as_float(((unsigned)s) << 16);
}

// ---------------------------------------------------------------------------
// Tiled bf16 GEMM, f32 accumulate. TRANSB=0: C = A @ B   (B is k x n row-major)
//                                  TRANSB=1: C = A @ B^T (B is m x k row-major)
// Block tile 128x128, K-slab 32, double-buffered LDS. 256 threads = 8 wave32
// (2x4 grid); each wave owns a 64x32 sub-tile = 4x2 fragments of 16x16.
// Tiles staged K-contiguous per lane: sA[buf][row][k], sBt[buf][col][k]
// -> every fragment load is two aligned 16-byte LDS reads (ds_load_b128).
// ---------------------------------------------------------------------------
#define BM 128
#define BN 128
#define BKK 32
#define TSTR 40   // padded K-stride (bf16 units); 80 bytes = multiple of 16B

template<int TRANSB, int OUTBF16>
__global__ __launch_bounds__(256) void k_gemm_bf16(
    const unsigned short* __restrict__ A,
    const unsigned short* __restrict__ B,
    void* __restrict__ Cout,
    int Ksz, int lda, int ldb, int ldc)
{
  __shared__ unsigned short sA [2][BM * TSTR];
  __shared__ unsigned short sBt[2][BN * TSTR];

  const int tid    = threadIdx.x;
  const int lane   = tid & 31;
  const int wid    = tid >> 5;
  const int wm     = wid >> 2;      // 0..1
  const int wn     = wid & 3;       // 0..3
  const int bm0    = blockIdx.y * BM;
  const int bn0    = blockIdx.x * BN;
  const int mlo    = lane & 15;
  const int hiHalf = lane >> 4;     // 0 or 1

  // stage one K-slab into LDS buffer sb (A tile + transposed B tile)
  auto stage = [&](int sb, int kk0) {
    // A tile: 128x32 bf16 = 512 x 16B, 2 chunks per thread
#pragma unroll
    for (int i = 0; i < 2; ++i) {
      int d   = i * 256 + tid;
      int row = d >> 2;           // 4 chunks per 32-elem row
      int q   = d & 3;
      copy16_g2l(A + (size_t)(bm0 + row) * lda + kk0 + 8 * q,
                 &sA[sb][row * TSTR + 8 * q]);
    }
    if (TRANSB) {
      // B is m x k row-major; B^T tile columns are B rows: contiguous-K copy
#pragma unroll
      for (int i = 0; i < 2; ++i) {
        int d  = i * 256 + tid;
        int nn = d >> 2;
        int q  = d & 3;
        copy16_g2l(B + (size_t)(bn0 + nn) * ldb + kk0 + 8 * q,
                   &sBt[sb][nn * TSTR + 8 * q]);
      }
    } else {
      // B is k x n row-major; read coalesced along n, scatter u16 into sBt
#pragma unroll
      for (int i = 0; i < 2; ++i) {
        int d  = i * 256 + tid;
        int kk = d >> 4;           // 16 chunks per 128-elem n-row
        int q  = d & 15;
        uint4 vB = *(const uint4*)(B + (size_t)(kk0 + kk) * ldb + bn0 + 8 * q);
        const unsigned* w = (const unsigned*)&vB;
#pragma unroll
        for (int e = 0; e < 8; ++e) {
          unsigned short val = (unsigned short)((e & 1) ? (w[e >> 1] >> 16)
                                                        : (w[e >> 1] & 0xFFFFu));
          sBt[sb][(8 * q + e) * TSTR + kk] = val;
        }
      }
    }
  };

  v8f acc[4][2];
#pragma unroll
  for (int i = 0; i < 4; ++i)
#pragma unroll
    for (int j = 0; j < 2; ++j)
#pragma unroll
      for (int r = 0; r < 8; ++r) acc[i][j][r] = 0.f;

  // prologue: first slab
  stage(0, 0);
  asyncs_wait();
  __syncthreads();

  int buf = 0;
  for (int k0 = 0; k0 < Ksz; k0 += BKK) {
    const int  nbuf    = buf ^ 1;
    const bool hasNext = (k0 + BKK) < Ksz;
    if (hasNext) stage(nbuf, k0 + BKK);   // overlaps with compute below

    // ---- fragments (CDNA5 16-bit WMMA layouts); all reads are b128
    union Frag { v16bf v; unsigned u[8]; } fa[4], fb[2];
    const int kbA = hiHalf ? 8 : 0;   // A: lanes 0-15 K{0-7,16-23}, 16-31 K{8-15,24-31}
    const int kbB = hiHalf ? 8 : 0;   // B: lanes 0-15 K{0-7,16-23}, 16-31 K{8-15,24-31}
#pragma unroll
    for (int fm = 0; fm < 4; ++fm) {
      const unsigned short* baseA = &sA[buf][(wm * 64 + fm * 16 + mlo) * TSTR];
      *(uint4*)&fa[fm].u[0] = *(const uint4*)(baseA + kbA);
      *(uint4*)&fa[fm].u[4] = *(const uint4*)(baseA + kbA + 16);
    }
#pragma unroll
    for (int fn = 0; fn < 2; ++fn) {
      const unsigned short* baseB = &sBt[buf][(wn * 32 + fn * 16 + mlo) * TSTR + kbB];
      *(uint4*)&fb[fn].u[0] = *(const uint4*)(baseB);
      *(uint4*)&fb[fn].u[4] = *(const uint4*)(baseB + 16);
    }

#pragma unroll
    for (int fm = 0; fm < 4; ++fm)
#pragma unroll
      for (int fn = 0; fn < 2; ++fn)
        acc[fm][fn] = __builtin_amdgcn_wmma_f32_16x16x32_bf16(
            false, fa[fm].v, false, fb[fn].v, (short)0, acc[fm][fn], false, false);

    if (hasNext) {
      asyncs_wait();     // next slab landed in LDS
      __syncthreads();   // visible to all waves; also fences this slab's reads
    }
    buf = nbuf;
  }

  // ---- epilogue: C/D layout — VGPR r, lanes 0-15: M=r, lanes 16-31: M=r+8
#pragma unroll
  for (int fm = 0; fm < 4; ++fm)
#pragma unroll
    for (int fn = 0; fn < 2; ++fn) {
      int colg = bn0 + wn * 32 + fn * 16 + mlo;
#pragma unroll
      for (int r = 0; r < 8; ++r) {
        int rowg = bm0 + wm * 64 + fm * 16 + r + hiHalf * 8;
        float val = acc[fm][fn][r];
        if (OUTBF16)
          ((unsigned short*)Cout)[(size_t)rowg * ldc + colg] = f2bf(val);
        else
          ((float*)Cout)[(size_t)rowg * ldc + colg] = val;
      }
    }
}

// ---------------------------------------------------------------------------
// Elementwise / reduction kernels
// ---------------------------------------------------------------------------
__global__ void k_f32_to_bf16(const float* __restrict__ s, unsigned short* __restrict__ d, int n){
  for (int i = blockIdx.x * blockDim.x + threadIdx.x; i < n; i += gridDim.x * blockDim.x)
    d[i] = f2bf(s[i]);
}

__global__ void k_fill_f32(float* p, float v, int n){
  for (int i = blockIdx.x * blockDim.x + threadIdx.x; i < n; i += gridDim.x * blockDim.x)
    p[i] = v;
}

__global__ void k_set_u32(unsigned* p, unsigned v){
  if (threadIdx.x == 0 && blockIdx.x == 0) *p = v;
}

__global__ __launch_bounds__(256) void k_rowsumsq_f32(const float* __restrict__ x, float* out, int cols){
  __shared__ float sm[256];
  int row = blockIdx.x;
  float s = 0.f;
  for (int j = threadIdx.x; j < cols; j += 256) { float v = x[(size_t)row * cols + j]; s += v * v; }
  sm[threadIdx.x] = s; __syncthreads();
  for (int o = 128; o; o >>= 1) { if (threadIdx.x < o) sm[threadIdx.x] += sm[threadIdx.x + o]; __syncthreads(); }
  if (!threadIdx.x) out[row] = sm[0];
}

__global__ __launch_bounds__(256) void k_rowsumsq_bf16(const unsigned short* __restrict__ x, float* out, int cols){
  __shared__ float sm[256];
  int row = blockIdx.x;
  float s = 0.f;
  for (int j = threadIdx.x; j < cols; j += 256) { float v = bf2f(x[(size_t)row * cols + j]); s += v * v; }
  sm[threadIdx.x] = s; __syncthreads();
  for (int o = 128; o; o >>= 1) { if (threadIdx.x < o) sm[threadIdx.x] += sm[threadIdx.x + o]; __syncthreads(); }
  if (!threadIdx.x) out[row] = sm[0];
}

// D[i,j] = sqrt(relu(x2a[i] + x2b[j] - 2*gram[i,j]))  -> bf16 (safe sqrt)
__global__ __launch_bounds__(256) void k_assemble_dist(
    const float* __restrict__ gram, const float* __restrict__ x2a,
    const float* __restrict__ x2b, unsigned short* __restrict__ outb, int m)
{
  int row = blockIdx.x; float xa = x2a[row];
  for (int j = threadIdx.x; j < m; j += 256) {
    float d2 = xa + x2b[j] - 2.f * gram[(size_t)row * m + j];
    d2 = d2 > 0.f ? d2 : 0.f;
    float dv = d2 > 0.f ? sqrtf(d2) : 0.f;
    outb[(size_t)row * m + j] = f2bf(dv);
  }
}

// C_W[i,j] = relu(...) (squared dist, f32) + running global max -> key
__global__ __launch_bounds__(256) void k_assemble_cw(
    const float* __restrict__ gram, const float* __restrict__ x2a,
    const float* __restrict__ x2b, float* __restrict__ out, unsigned* key, int m)
{
  __shared__ float sm[256];
  int row = blockIdx.x; float xa = x2a[row];
  float mx = -INFINITY;
  for (int j = threadIdx.x; j < m; j += 256) {
    float d2 = xa + x2b[j] - 2.f * gram[(size_t)row * m + j];
    d2 = d2 > 0.f ? d2 : 0.f;
    out[(size_t)row * m + j] = d2;
    mx = fmaxf(mx, d2);
  }
  sm[threadIdx.x] = mx; __syncthreads();
  for (int o = 128; o; o >>= 1) { if (threadIdx.x < o) sm[threadIdx.x] = fmaxf(sm[threadIdx.x], sm[threadIdx.x + o]); __syncthreads(); }
  if (!threadIdx.x) atomicMax(key, fkey(sm[0]));
}

__global__ __launch_bounds__(256) void k_rowsum(const float* __restrict__ P, float* mu, int m){
  __shared__ float sm[256];
  int row = blockIdx.x;
  float s = 0.f;
  for (int j = threadIdx.x; j < m; j += 256) s += P[(size_t)row * m + j];
  sm[threadIdx.x] = s; __syncthreads();
  for (int o = 128; o; o >>= 1) { if (threadIdx.x < o) sm[threadIdx.x] += sm[threadIdx.x + o]; __syncthreads(); }
  if (!threadIdx.x) mu[row] = sm[0];
}

__global__ __launch_bounds__(256) void k_colsum(const float* __restrict__ P, float* nu, int n, int m){
  int col = blockIdx.x * 256 + threadIdx.x;
  float s = 0.f;
  for (int i = 0; i < n; ++i) s += P[(size_t)i * m + col];
  nu[col] = s;
}

// C_GW = rgen[i]*nu[j] + mu[i]*creal[j] - 2*T2[i,j]  (in place over T2)
__global__ __launch_bounds__(256) void k_combine_gw(
    float* __restrict__ cgw, const float* __restrict__ rgen, const float* __restrict__ creal,
    const float* __restrict__ mu, const float* __restrict__ nu, int m)
{
  int row = blockIdx.x; float rg = rgen[row], mr = mu[row];
  for (int j = threadIdx.x; j < m; j += 256) {
    size_t idx = (size_t)row * m + j;
    cgw[idx] = rg * nu[j] + mr * creal[j] - 2.f * cgw[idx];
  }
}

__global__ __launch_bounds__(256) void k_maxred(const float* __restrict__ x, unsigned* key, int n, int useAbs){
  __shared__ float sm[256];
  float mx = -INFINITY;
  for (int i = blockIdx.x * 256 + threadIdx.x; i < n; i += gridDim.x * 256) {
    float v = x[i]; if (useAbs) v = fabsf(v);
    mx = fmaxf(mx, v);
  }
  sm[threadIdx.x] = mx; __syncthreads();
  for (int o = 128; o; o >>= 1) { if (threadIdx.x < o) sm[threadIdx.x] = fmaxf(sm[threadIdx.x], sm[threadIdx.x + o]); __syncthreads(); }
  if (!threadIdx.x) atomicMax(key, fkey(sm[0]));
}

// C_fused = (1-LAM)*C_W/cwmax + LAM*C_GW/gmax (with >0 guards)
__global__ void k_fused(const float* __restrict__ cw, const float* __restrict__ cgw,
                        float* __restrict__ out, const unsigned* __restrict__ keys, int n){
  float cwmax = funkey(keys[0]), gmax = funkey(keys[1]);
  float ic = cwmax > 0.f ? 1.f / cwmax : 1.f;
  float ig = gmax  > 0.f ? 1.f / gmax  : 1.f;
  for (int i = blockIdx.x * blockDim.x + threadIdx.x; i < n; i += gridDim.x * blockDim.x)
    out[i] = (1.f - LAMW) * (cw[i] * ic) + LAMW * (cgw[i] * ig);
}

// log_K = -(C / (cmax+1e-8)) / eps   (cmax>0 guard), in place
__global__ void k_logk(float* __restrict__ x, const unsigned* __restrict__ keys, int n){
  float cmax = funkey(keys[2]);
  float inv = cmax > 0.f ? 1.f / (cmax + 1e-8f) : 1.f;
  float sc = -inv / EPSREG;
  for (int i = blockIdx.x * blockDim.x + threadIdx.x; i < n; i += gridDim.x * blockDim.x)
    x[i] = x[i] * sc;
}

// u[i] = log_a - logsumexp_j(log_K[i,j] + v[j])   — block per row, online LSE
__global__ __launch_bounds__(256) void k_row_lse(
    const float* __restrict__ K, const float* __restrict__ v, float* __restrict__ u, int m)
{
  __shared__ float smx[256], ssm[256];
  int row = blockIdx.x;
  const float* Kr = K + (size_t)row * m;
  float mx = -INFINITY, s = 0.f;
  for (int j = threadIdx.x; j < m; j += 256) {
    float val = Kr[j] + v[j];
    if (val > mx) { s = s * expf(mx - val) + 1.f; mx = val; }
    else          { s += expf(val - mx); }
  }
  smx[threadIdx.x] = mx; ssm[threadIdx.x] = s; __syncthreads();
  for (int o = 128; o; o >>= 1) {
    if (threadIdx.x < o) {
      float m1 = smx[threadIdx.x], s1 = ssm[threadIdx.x];
      float m2 = smx[threadIdx.x + o], s2 = ssm[threadIdx.x + o];
      float M = fmaxf(m1, m2);
      ssm[threadIdx.x] = s1 * expf(m1 - M) + s2 * expf(m2 - M);
      smx[threadIdx.x] = M;
    }
    __syncthreads();
  }
  if (!threadIdx.x) u[row] = LOG_AB - (smx[0] + logf(ssm[0]));
}

// v[j] = log_b - logsumexp_i(log_K[i,j] + u[i])   — thread per column (coalesced)
__global__ __launch_bounds__(256) void k_col_lse(
    const float* __restrict__ K, const float* __restrict__ u, float* __restrict__ v, int n, int m)
{
  int col = blockIdx.x * 256 + threadIdx.x;
  float mx = -INFINITY, s = 0.f;
  for (int i = 0; i < n; ++i) {
    float val = K[(size_t)i * m + col] + u[i];
    if (val > mx) { s = s * expf(mx - val) + 1.f; mx = val; }
    else          { s += expf(val - mx); }
  }
  v[col] = LOG_AB - (mx + logf(s));
}

// P = exp(u_i + log_K + v_j); accumulate total into *psum
__global__ __launch_bounds__(256) void k_form_p(
    const float* __restrict__ K, const float* __restrict__ u, const float* __restrict__ v,
    float* __restrict__ P, float* psum, int m)
{
  __shared__ float sm[256];
  int row = blockIdx.x; float ur = u[row];
  float s = 0.f;
  for (int j = threadIdx.x; j < m; j += 256) {
    float p = expf(ur + K[(size_t)row * m + j] + v[j]);
    P[(size_t)row * m + j] = p;
    s += p;
  }
  sm[threadIdx.x] = s; __syncthreads();
  for (int o = 128; o; o >>= 1) { if (threadIdx.x < o) sm[threadIdx.x] += sm[threadIdx.x + o]; __syncthreads(); }
  if (!threadIdx.x) atomicAdd(psum, sm[0]);
}

__global__ void k_scale_p(float* __restrict__ P, const float* __restrict__ psum, int n){
  float sc = 1.0f / (*psum + 1e-10f);   // mass = min(sum a, sum b) = 1
  for (int i = blockIdx.x * blockDim.x + threadIdx.x; i < n; i += gridDim.x * blockDim.x)
    P[i] *= sc;
}

// C_final = (1-LAM)*C_W + LAM*C_GW   (in place over C_W)
__global__ void k_final(float* __restrict__ cw, const float* __restrict__ cgw, int n){
  for (int i = blockIdx.x * blockDim.x + threadIdx.x; i < n; i += gridDim.x * blockDim.x)
    cw[i] = (1.f - LAMW) * cw[i] + LAMW * cgw[i];
}

// ---------------------------------------------------------------------------
// Orchestration
// ---------------------------------------------------------------------------
extern "C" void kernel_launch(void* const* d_in, const int* in_sizes, int n_in,
                              void* d_out, int out_size, void* d_ws, size_t ws_size,
                              hipStream_t stream)
{
  (void)in_sizes; (void)n_in; (void)out_size; (void)ws_size;
  const float* Fg = (const float*)d_in[0];
  const float* Fr = (const float*)d_in[1];

  float* P  = (float*)d_out;                       // coupling (output 1)
  float* CW = (float*)d_out + (size_t)NM_TOT;      // C_W, becomes C_final (output 2)

  char* ws = (char*)d_ws;
  float*          bufF   = (float*)ws;                                  // 64 MB: gram / C_fused / log_K
  float*          cgw    = (float*)(ws + ((size_t)64  << 20));          // 64 MB: C_GW
  unsigned short* dgenb  = (unsigned short*)(ws + ((size_t)128 << 20)); // 32 MB
  unsigned short* drealb = (unsigned short*)(ws + ((size_t)160 << 20)); // 32 MB
  unsigned short* pb     = (unsigned short*)(ws + ((size_t)192 << 20)); // 32 MB
  unsigned short* tb     = (unsigned short*)(ws + ((size_t)224 << 20)); // 32 MB
  unsigned short* fgb    = (unsigned short*)(ws + ((size_t)256 << 20)); // 2 MB
  unsigned short* frb    = (unsigned short*)(ws + ((size_t)258 << 20)); // 2 MB
  float* x2g   = (float*)(ws + ((size_t)260 << 20));
  float* x2r   = x2g   + N4K;
  float* rgen  = x2r   + N4K;
  float* creal = rgen  + N4K;
  float* mu    = creal + N4K;
  float* nu    = mu    + N4K;
  float* uu    = nu    + N4K;
  float* vv    = uu    + N4K;
  unsigned* keys = (unsigned*)(vv + N4K);   // [0]=cwmax [1]=gmax [2]=cmax [3]=psum

  dim3 b256(256);
  dim3 gNM(4096);       // grid-stride over 16.7M
  dim3 gRows(N4K);      // block per row
  dim3 gCols(N4K/256);  // thread per column
  dim3 gGemm(N4K/BN, N4K/BM);
  dim3 gFeat(2048);

  // --- precompute: bf16 features, squared norms
  k_f32_to_bf16<<<gFeat, b256, 0, stream>>>(Fg, fgb, N4K * DFEAT);
  k_f32_to_bf16<<<gFeat, b256, 0, stream>>>(Fr, frb, N4K * DFEAT);
  k_rowsumsq_f32<<<gRows, b256, 0, stream>>>(Fg, x2g, DFEAT);
  k_rowsumsq_f32<<<gRows, b256, 0, stream>>>(Fr, x2r, DFEAT);

  // --- D_gen = cdist(Fg,Fg): gram (WMMA) -> assemble -> bf16; row sums of squares
  k_gemm_bf16<1,0><<<gGemm, b256, 0, stream>>>(fgb, fgb, bufF, DFEAT, DFEAT, DFEAT, N4K);
  k_assemble_dist<<<gRows, b256, 0, stream>>>(bufF, x2g, x2g, dgenb, N4K);
  k_rowsumsq_bf16<<<gRows, b256, 0, stream>>>(dgenb, rgen, N4K);

  // --- D_real (symmetric)
  k_gemm_bf16<1,0><<<gGemm, b256, 0, stream>>>(frb, frb, bufF, DFEAT, DFEAT, DFEAT, N4K);
  k_assemble_dist<<<gRows, b256, 0, stream>>>(bufF, x2r, x2r, drealb, N4K);
  k_rowsumsq_bf16<<<gRows, b256, 0, stream>>>(drealb, creal, N4K);

  // --- C_W = sq_dists(Fg,Fr) (f32, kept for final), cwmax
  k_gemm_bf16<1,0><<<gGemm, b256, 0, stream>>>(fgb, frb, bufF, DFEAT, DFEAT, DFEAT, N4K);
  k_set_u32<<<1, 1, 0, stream>>>(keys + 0, KEY_NEG_INF);
  k_assemble_cw<<<gRows, b256, 0, stream>>>(bufF, x2g, x2r, CW, keys + 0, N4K);

  // --- coupling init: a b^T = 1/(N*M)
  k_fill_f32<<<gNM, b256, 0, stream>>>(P, 1.0f / ((float)N4K * (float)N4K), NM_TOT);

  for (int t = 0; t < 5; ++t) {
    // structure cost from current coupling
    k_f32_to_bf16<<<gNM, b256, 0, stream>>>(P, pb, NM_TOT);
    k_rowsum<<<gRows, b256, 0, stream>>>(P, mu, N4K);
    k_colsum<<<gCols, b256, 0, stream>>>(P, nu, N4K, N4K);
    // T = D_gen @ P  (bf16 out), then C3 = T @ D_real^T (symmetric D_real)
    k_gemm_bf16<0,1><<<gGemm, b256, 0, stream>>>(dgenb, pb, tb, N4K, N4K, N4K, N4K);
    k_gemm_bf16<1,0><<<gGemm, b256, 0, stream>>>(tb, drealb, cgw, N4K, N4K, N4K, N4K);
    k_combine_gw<<<gRows, b256, 0, stream>>>(cgw, rgen, creal, mu, nu, N4K);

    // fused cost and log-kernel
    k_set_u32<<<1, 1, 0, stream>>>(keys + 1, KEY_NEG_INF);
    k_maxred<<<gNM, b256, 0, stream>>>(cgw, keys + 1, NM_TOT, 1);
    k_fused<<<gNM, b256, 0, stream>>>(CW, cgw, bufF, keys, NM_TOT);
    k_set_u32<<<1, 1, 0, stream>>>(keys + 2, KEY_NEG_INF);
    k_maxred<<<gNM, b256, 0, stream>>>(bufF, keys + 2, NM_TOT, 0);
    k_logk<<<gNM, b256, 0, stream>>>(bufF, keys + 2, NM_TOT);

    // Sinkhorn (30 fixed iterations)
    k_fill_f32<<<16, b256, 0, stream>>>(uu, 0.f, N4K);
    k_fill_f32<<<16, b256, 0, stream>>>(vv, 0.f, N4K);
    for (int s = 0; s < 30; ++s) {
      k_row_lse<<<gRows, b256, 0, stream>>>(bufF, vv, uu, N4K);
      k_col_lse<<<gCols, b256, 0, stream>>>(bufF, uu, vv, N4K, N4K);
    }
    k_set_u32<<<1, 1, 0, stream>>>(keys + 3, 0u);  // psum = 0.0f
    k_form_p<<<gRows, b256, 0, stream>>>(bufF, uu, vv, P, (float*)(keys + 3), N4K);
    k_scale_p<<<gNM, b256, 0, stream>>>(P, (float*)(keys + 3), NM_TOT);
  }

  // C_final = (1-LAM)*C_W + LAM*C_GW  (in place in d_out second half)
  k_final<<<gNM, b256, 0, stream>>>(CW, cgw, NM_TOT);
}
